// GCL_52192442581787
// MI455X (gfx1250) — matrix operand order repaired and checked by reading, not compiled, not run on hardware
//
#include <hip/hip_runtime.h>
#include <hip/hip_bf16.h>

// ---------------------------------------------------------------------------
// GNN message-passing layer for MI455X (gfx1250, wave32, WMMA).
// Edge MLP + scatter-add + node MLP, all matmuls via v_wmma_f32_16x16x32_bf16.
// Weights staged in LDS per block; A/B operands software-pipelined.
// ---------------------------------------------------------------------------

#define N_NODES 10000
#define N_EDGES 640000
#define NF      128
#define HID     128
#define INV_NORM 0.01f   // 1/NORM_FACTOR

// LDS partition (in ushorts): 32768 (w1, 64KB) + 16384 (w2, 32KB) + 16384 (h1, 32KB)
#define LDS_W1_OFF 0
#define LDS_W2_OFF 32768
#define LDS_H1_OFF 49152
#define LDS_BYTES  131072

typedef __attribute__((ext_vector_type(16))) __bf16        v16bf;
typedef __attribute__((ext_vector_type(8)))  float         v8f;
typedef __attribute__((ext_vector_type(8)))  unsigned int  v8u;

// fp32 -> bf16 round-to-nearest-even (bit trick)
__device__ __forceinline__ unsigned short f2bf(float f) {
    unsigned int u = __float_as_uint(f);
    u = u + 0x7FFFu + ((u >> 16) & 1u);
    return (unsigned short)(u >> 16);
}

__device__ __forceinline__ float silu(float x) {
    return x / (1.0f + __expf(-x));
}

// Load a 16-element bf16 per-lane operand as two contiguous 16B chunks.
// Works for global and LDS pointers (addrspace inferred after inlining).
__device__ __forceinline__ v16bf load_bf16x16(const unsigned short* p0,
                                              const unsigned short* p1) {
    uint4 x = *(const uint4*)p0;
    uint4 y = *(const uint4*)p1;
    v8u u = {x.x, x.y, x.z, x.w, y.x, y.y, y.z, y.w};
    return __builtin_bit_cast(v16bf, u);
}

// B-operand tile fetch from packed layout (one 32B per-lane contiguous read).
__device__ __forceinline__ v16bf ldB(const unsigned short* base, int tile, int lane) {
    const unsigned short* p = base + (tile << 9) + lane * 16;
    return load_bf16x16(p, p + 8);
}

// Pack 16 fp32 values (two 8-float chunks) into a bf16 A-operand, with scale.
__device__ __forceinline__ v16bf pack_f32_to_bf16(const float* p0,
                                                  const float* p1,
                                                  float scale) {
    float4 a0 = *(const float4*)(p0);
    float4 a1 = *(const float4*)(p0 + 4);
    float4 b0 = *(const float4*)(p1);
    float4 b1 = *(const float4*)(p1 + 4);
    float f[16] = {a0.x, a0.y, a0.z, a0.w, a1.x, a1.y, a1.z, a1.w,
                   b0.x, b0.y, b0.z, b0.w, b1.x, b1.y, b1.z, b1.w};
    v8u u;
#pragma unroll
    for (int i = 0; i < 8; ++i) {
        unsigned int lo = f2bf(f[2 * i] * scale);
        unsigned int hi = f2bf(f[2 * i + 1] * scale);
        u[i] = lo | (hi << 16);
    }
    return __builtin_bit_cast(v16bf, u);
}

// ---------------------------------------------------------------------------
// Prep kernels
// ---------------------------------------------------------------------------

__global__ void convert_h_bf16_kernel(const float* __restrict__ h,
                                      unsigned short* __restrict__ hb, int n) {
    int i = blockIdx.x * blockDim.x + threadIdx.x;
    if (i < n) hb[i] = f2bf(h[i]);
}

__global__ void zero_f32_kernel(float* __restrict__ p, int n) {
    int i = blockIdx.x * blockDim.x + threadIdx.x;
    if (i < n) p[i] = 0.0f;
}

// Pack weight W[K][N] (row-major fp32) into the wave32 WMMA B-operand layout:
// per 32x16 (K x N) tile, 512 bf16 stored lane-major: lane l holds column
// (l&15), k 8h..8h+7 then 16+8h..16+8h+7 (h = l>>4), 16 bf16 packed 2-per-dword.
__global__ void pack_weight_kernel(const float* __restrict__ W,
                                   unsigned short* __restrict__ out,
                                   int K, int N) {
    int idx = blockIdx.x * blockDim.x + threadIdx.x;
    if (idx >= K * N) return;
    int tile   = idx >> 9;
    int within = idx & 511;
    int lane   = within >> 4;
    int q      = within & 15;
    int ntiles = N >> 4;
    int kt = tile / ntiles;
    int nt = tile % ntiles;
    int n  = nt * 16 + (lane & 15);
    int k  = kt * 32 + (lane >> 4) * 8 + (q & 7) + (q >> 3) * 16;
    out[idx] = f2bf(W[k * N + n]);
}

// Cooperative global -> LDS weight staging: 96KB as b128 copies.
__device__ __forceinline__ void stage_weights(unsigned short* smem,
                                              const unsigned short* w1p,
                                              const unsigned short* w2p,
                                              int tid) {
    const uint4* s1 = (const uint4*)w1p;
    uint4*       d1 = (uint4*)(smem + LDS_W1_OFF);
#pragma unroll
    for (int i = 0; i < 16; ++i) d1[tid + 256 * i] = s1[tid + 256 * i];   // 64KB
    const uint4* s2 = (const uint4*)w2p;
    uint4*       d2 = (uint4*)(smem + LDS_W2_OFF);
#pragma unroll
    for (int i = 0; i < 8; ++i) d2[tid + 256 * i] = s2[tid + 256 * i];    // 32KB
}

// ---------------------------------------------------------------------------
// Edge MLP: x = [h[row], h[col]]; mij = silu(silu(x@ew1+eb1)@ew2+eb2)
// plus atomic scatter-add of mij into agg[row].
// One wave = 16 edges; 8 waves/block => 128 edges/block; 5000 blocks.
// ---------------------------------------------------------------------------
__global__ __launch_bounds__(256)
void edge_mlp_kernel(const unsigned short* __restrict__ h_bf,
                     const long long* __restrict__ ei,      // [2, E] int64
                     const unsigned short* __restrict__ ew1p,
                     const float* __restrict__ eb1,
                     const unsigned short* __restrict__ ew2p,
                     const float* __restrict__ eb2,
                     float* __restrict__ mij_out,
                     float* __restrict__ agg) {
    extern __shared__ unsigned short smem[];
    unsigned short* w1  = smem + LDS_W1_OFF;   // ew1 packed, 64 tiles
    unsigned short* w2  = smem + LDS_W2_OFF;   // ew2 packed, 32 tiles
    unsigned short* h1s = smem + LDS_H1_OFF;   // per-wave 16x128 activations

    const int tid  = threadIdx.x;
    const int lane = tid & 31;
    const int wave = tid >> 5;
    const int m    = lane & 15;   // A-operand row / C-operand column
    const int half = lane >> 4;   // K-half selector
    const int eBase = blockIdx.x * 128 + wave * 16;

    stage_weights(smem, ew1p, ew2p, tid);
    __syncthreads();

    const long long ri = ei[eBase + m];
    const long long ci = ei[N_EDGES + eBase + m];
    const unsigned short* hr = h_bf + (size_t)ri * NF;
    const unsigned short* hc = h_bf + (size_t)ci * NF;

    const int off0 = half * 8;        // A layout: lanes<16 -> K0..7 / K16..23
    const int off1 = 16 + half * 8;   //           lanes>=16 -> K8..15 / K24..31

    auto ldA1 = [&](int kt) {
        const unsigned short* src = (kt < 4 ? hr : hc) + (kt & 3) * 32;
        return load_bf16x16(src + off0, src + off1);
    };

    const v8f vzero = {0.f, 0.f, 0.f, 0.f, 0.f, 0.f, 0.f, 0.f};
    v8f c[8];
#pragma unroll
    for (int nt = 0; nt < 8; ++nt) c[nt] = vzero;

    // ---- layer 1: [16 x 256] @ [256 x 128]; A from global gather, B from LDS,
    //      both double-buffered so loads overlap WMMA issue ----
    {
        v16bf A = ldA1(0);
        v16bf B = ldB(w1, 0, lane);
#pragma unroll
        for (int kt = 0; kt < 8; ++kt) {
            v16bf An = (kt < 7) ? ldA1(kt + 1) : A;
#pragma unroll
            for (int nt = 0; nt < 8; ++nt) {
                int t = kt * 8 + nt;
                v16bf Bn = (t < 63) ? ldB(w1, t + 1, lane) : B;
                c[nt] = __builtin_amdgcn_wmma_f32_16x16x32_bf16(
                    false, A, false, B, (short)0, c[nt], false, false);
                B = Bn;
            }
            A = An;
        }
    }

    // bias + SiLU -> per-wave LDS tile (becomes layer-2 A operand)
    unsigned short* h1 = h1s + wave * 16 * HID;
    const int n = m;
#pragma unroll
    for (int nt = 0; nt < 8; ++nt) {
        float b = eb1[nt * 16 + n];
#pragma unroll
        for (int j = 0; j < 8; ++j) {
            int r = half * 8 + j;   // C layout: vgpr j, lanes>=16 -> rows 8..15
            h1[r * HID + nt * 16 + n] = f2bf(silu(c[nt][j] + b));
        }
    }

    // ---- layer 2: [16 x 128] @ [128 x 128]; A and B both from LDS ----
    v8f d[8];
#pragma unroll
    for (int nt = 0; nt < 8; ++nt) d[nt] = vzero;
    {
        auto ldA2 = [&](int kt) {
            const unsigned short* src = h1 + m * HID + kt * 32;
            return load_bf16x16(src + off0, src + off1);
        };
        v16bf A = ldA2(0);
        v16bf B = ldB(w2, 0, lane);
#pragma unroll
        for (int kt = 0; kt < 4; ++kt) {
            v16bf An = (kt < 3) ? ldA2(kt + 1) : A;
#pragma unroll
            for (int nt = 0; nt < 8; ++nt) {
                int t = kt * 8 + nt;
                v16bf Bn = (t < 31) ? ldB(w2, t + 1, lane) : B;
                d[nt] = __builtin_amdgcn_wmma_f32_16x16x32_bf16(
                    false, A, false, B, (short)0, d[nt], false, false);
                B = Bn;
            }
            A = An;
        }
    }

    // bias + SiLU -> mij (fp32 output) + atomic scatter-add into agg[row]
    const int rii = (int)ri;  // node index fits in 32 bits
#pragma unroll
    for (int nt = 0; nt < 8; ++nt) {
        float b = eb2[nt * 16 + n];
#pragma unroll
        for (int j = 0; j < 8; ++j) {
            int r = half * 8 + j;
            float v = silu(d[nt][j] + b);
            mij_out[(size_t)(eBase + r) * HID + nt * 16 + n] = v;
            int rr = __shfl(rii, r, 32);  // row index of edge eBase+r
            atomicAdd(&agg[(size_t)rr * HID + nt * 16 + n], v);
        }
    }
}

// ---------------------------------------------------------------------------
// Node MLP: a = [h, agg/100]; out = h + silu(a@nw1+nb1)@nw2 + nb2
// One wave = 16 nodes; 625 tiles -> 79 blocks of 8 waves (tail waves exit).
// ---------------------------------------------------------------------------
__global__ __launch_bounds__(256)
void node_mlp_kernel(const float* __restrict__ h_f32,
                     const unsigned short* __restrict__ h_bf,
                     const float* __restrict__ agg,
                     const unsigned short* __restrict__ nw1p,
                     const float* __restrict__ nb1,
                     const unsigned short* __restrict__ nw2p,
                     const float* __restrict__ nb2,
                     float* __restrict__ out) {
    extern __shared__ unsigned short smem[];
    unsigned short* w1  = smem + LDS_W1_OFF;
    unsigned short* w2  = smem + LDS_W2_OFF;
    unsigned short* h1s = smem + LDS_H1_OFF;

    const int tid  = threadIdx.x;
    const int lane = tid & 31;
    const int wave = tid >> 5;

    stage_weights(smem, nw1p, nw2p, tid);
    __syncthreads();

    const int tile  = blockIdx.x * 8 + wave;
    const int vBase = tile * 16;
    if (vBase >= N_NODES) return;   // uniform per wave: EXEC stays all-ones

    const int m    = lane & 15;
    const int half = lane >> 4;
    const int n    = m;
    const int off0 = half * 8;
    const int off1 = 16 + half * 8;

    const unsigned short* hb = h_bf + (size_t)(vBase + m) * NF;
    const float*          ag = agg + (size_t)(vBase + m) * HID;

    auto ldA1 = [&](int kt) {
        if (kt < 4) {
            const unsigned short* src = hb + kt * 32;
            return load_bf16x16(src + off0, src + off1);
        } else {
            const float* src = ag + (kt - 4) * 32;
            return pack_f32_to_bf16(src + off0, src + off1, INV_NORM);
        }
    };

    const v8f vzero = {0.f, 0.f, 0.f, 0.f, 0.f, 0.f, 0.f, 0.f};
    v8f c[8];
#pragma unroll
    for (int nt = 0; nt < 8; ++nt) c[nt] = vzero;

    // ---- layer 1: [16 x 256] @ nw1 (first half h, second half agg/100) ----
    {
        v16bf A = ldA1(0);
        v16bf B = ldB(w1, 0, lane);
#pragma unroll
        for (int kt = 0; kt < 8; ++kt) {
            v16bf An = (kt < 7) ? ldA1(kt + 1) : A;
#pragma unroll
            for (int nt = 0; nt < 8; ++nt) {
                int t = kt * 8 + nt;
                v16bf Bn = (t < 63) ? ldB(w1, t + 1, lane) : B;
                c[nt] = __builtin_amdgcn_wmma_f32_16x16x32_bf16(
                    false, A, false, B, (short)0, c[nt], false, false);
                B = Bn;
            }
            A = An;
        }
    }

    unsigned short* h1 = h1s + wave * 16 * HID;
#pragma unroll
    for (int nt = 0; nt < 8; ++nt) {
        float b = nb1[nt * 16 + n];
#pragma unroll
        for (int j = 0; j < 8; ++j) {
            int r = half * 8 + j;
            h1[r * HID + nt * 16 + n] = f2bf(silu(c[nt][j] + b));
        }
    }

    // ---- layer 2: [16 x 128] @ nw2 ----
    v8f d[8];
#pragma unroll
    for (int nt = 0; nt < 8; ++nt) d[nt] = vzero;
    {
        auto ldA2 = [&](int kt) {
            const unsigned short* src = h1 + m * HID + kt * 32;
            return load_bf16x16(src + off0, src + off1);
        };
        v16bf A = ldA2(0);
        v16bf B = ldB(w2, 0, lane);
#pragma unroll
        for (int kt = 0; kt < 4; ++kt) {
            v16bf An = (kt < 3) ? ldA2(kt + 1) : A;
#pragma unroll
            for (int nt = 0; nt < 8; ++nt) {
                int t = kt * 8 + nt;
                v16bf Bn = (t < 31) ? ldB(w2, t + 1, lane) : B;
                d[nt] = __builtin_amdgcn_wmma_f32_16x16x32_bf16(
                    false, A, false, B, (short)0, d[nt], false, false);
                B = Bn;
            }
            A = An;
        }
    }

    // residual + bias, store out (fp32)
#pragma unroll
    for (int nt = 0; nt < 8; ++nt) {
        float b = nb2[nt * 16 + n];
#pragma unroll
        for (int j = 0; j < 8; ++j) {
            int r = half * 8 + j;
            size_t o = (size_t)(vBase + r) * NF + nt * 16 + n;
            out[o] = h_f32[o] + d[nt][j] + b;
        }
    }
}

// ---------------------------------------------------------------------------
// Launch
// ---------------------------------------------------------------------------
extern "C" void kernel_launch(void* const* d_in, const int* in_sizes, int n_in,
                              void* d_out, int out_size, void* d_ws, size_t ws_size,
                              hipStream_t stream) {
    (void)in_sizes; (void)n_in; (void)out_size; (void)ws_size;

    const float*     h   = (const float*)d_in[0];
    const long long* ei  = (const long long*)d_in[1];  // int64 [2, E]
    const float*     ew1 = (const float*)d_in[2];
    const float*     eb1 = (const float*)d_in[3];
    const float*     ew2 = (const float*)d_in[4];
    const float*     eb2 = (const float*)d_in[5];
    const float*     nw1 = (const float*)d_in[6];
    const float*     nb1 = (const float*)d_in[7];
    const float*     nw2 = (const float*)d_in[8];
    const float*     nb2 = (const float*)d_in[9];

    float* out = (float*)d_out;              // [V*NF]
    float* mij = out + (size_t)N_NODES * NF; // [E*HID]

    // Workspace layout (~7.9 MB total, all offsets 256B-aligned):
    char* ws = (char*)d_ws;
    unsigned short* h_bf = (unsigned short*)(ws);                 // 2,560,000 B
    unsigned short* ew1p = (unsigned short*)(ws + 2560000);       //    65,536 B
    unsigned short* ew2p = (unsigned short*)(ws + 2625536);       //    32,768 B
    unsigned short* nw1p = (unsigned short*)(ws + 2658304);       //    65,536 B
    unsigned short* nw2p = (unsigned short*)(ws + 2723840);       //    32,768 B
    float*          agg  = (float*)(ws + 2756608);                // 5,120,000 B

    // Allow 128KB dynamic LDS (gfx1250 WGP has 320KB). Host-side attribute set,
    // not a stream operation -> safe under graph capture.
    static bool attr_done = false;
    if (!attr_done) {
        hipFuncSetAttribute((const void*)edge_mlp_kernel,
                            hipFuncAttributeMaxDynamicSharedMemorySize, LDS_BYTES);
        hipFuncSetAttribute((const void*)node_mlp_kernel,
                            hipFuncAttributeMaxDynamicSharedMemorySize, LDS_BYTES);
        attr_done = true;
    }

    const int hEl = N_NODES * NF;  // 1,280,000

    convert_h_bf16_kernel<<<(hEl + 255) / 256, 256, 0, stream>>>(h, h_bf, hEl);
    pack_weight_kernel<<<(2 * NF * HID + 255) / 256, 256, 0, stream>>>(ew1, ew1p, 2 * NF, HID);
    pack_weight_kernel<<<(HID * HID + 255) / 256, 256, 0, stream>>>(ew2, ew2p, HID, HID);
    pack_weight_kernel<<<(2 * NF * HID + 255) / 256, 256, 0, stream>>>(nw1, nw1p, HID + NF, HID);
    pack_weight_kernel<<<(HID * HID + 255) / 256, 256, 0, stream>>>(nw2, nw2p, HID, NF);
    zero_f32_kernel<<<(N_NODES * HID + 255) / 256, 256, 0, stream>>>(agg, N_NODES * HID);

    edge_mlp_kernel<<<N_EDGES / 128, 256, LDS_BYTES, stream>>>(
        h_bf, ei, ew1p, eb1, ew2p, eb2, mij, agg);

    const int nodeTiles = N_NODES / 16;                    // 625
    node_mlp_kernel<<<(nodeTiles + 7) / 8, 256, LDS_BYTES, stream>>>(
        h, h_bf, agg, nw1p, nb1, nw2p, nb2, out);
}
